// VisionCompressor_58110907515283
// MI455X (gfx1250) — matrix operand order repaired and checked by reading, not compiled, not run on hardware
//
#include <hip/hip_runtime.h>

// ---------------------------------------------------------------- types
typedef __attribute__((ext_vector_type(16))) _Float16 v16h;
typedef __attribute__((ext_vector_type(8)))  _Float16 v8h;
typedef __attribute__((ext_vector_type(8)))  float    v8f;

// ---------------------------------------------------------------- constants
constexpr int   BQ    = 2;
constexpr int   NTOK  = 768;
constexpr int   DDIM  = 2048;
constexpr int   CLIPC = 1024;
constexpr int   LLAY  = 25;
constexpr int   JJ    = 7;
constexpr int   HP    = 24;
constexpr int   HWN   = 576;     // HP*HP
constexpr int   KPAT  = 588;     // 3*14*14
constexpr int   KPATP = 608;     // padded to multiple of 32
constexpr float SCALE = 0.03125f;   // 1024^-0.5
constexpr float LN_EPS = 1e-5f;
constexpr float COS_EPS = 1e-8f;

// ---------------------------------------------------------------- find image token
__global__ __launch_bounds__(256) void k_find_idx(const int* __restrict__ ids,
                                                  int* __restrict__ idxout) {
  __shared__ int s[BQ];
  if (threadIdx.x < BQ) s[threadIdx.x] = NTOK;
  __syncthreads();
  for (int t = threadIdx.x; t < BQ * NTOK; t += blockDim.x) {
    if (ids[t] == -200) atomicMin(&s[t / NTOK], t % NTOK);
  }
  __syncthreads();
  if (threadIdx.x < BQ) idxout[threadIdx.x] = s[threadIdx.x];
}

// ---------------------------------------------------------------- patch extraction -> f16 (zero padded K)
__global__ __launch_bounds__(256) void k_patchify(const float* __restrict__ images,
                                                  _Float16* __restrict__ patchesH) {
  long long t = (long long)blockIdx.x * blockDim.x + threadIdx.x;
  long long total = (long long)BQ * HWN * KPATP;
  if (t >= total) return;
  int kk = (int)(t % KPATP);
  int p  = (int)((t / KPATP) % HWN);
  int b  = (int)(t / ((long long)KPATP * HWN));
  _Float16 v = (_Float16)0.0f;
  if (kk < KPAT) {
    int c = kk / 196, rem = kk % 196, i = rem / 14, j = rem % 14;
    int py = p / HP, px = p % HP;
    v = (_Float16)images[(((long long)b * 4 + c) * 336 + (py * 14 + i)) * 336 + (px * 14 + j)];
  }
  patchesH[t] = v;
}

// ---------------------------------------------------------------- f32 -> f16 convert with K padding
__global__ __launch_bounds__(256) void k_cvt_pad(const float* __restrict__ src,
                                                 _Float16* __restrict__ dst,
                                                 long long rows, int ksrc, int kdst) {
  long long t = (long long)blockIdx.x * blockDim.x + threadIdx.x;
  if (t >= rows * (long long)kdst) return;
  long long r = t / kdst;
  int k = (int)(t % kdst);
  dst[t] = (k < ksrc) ? (_Float16)src[r * ksrc + k] : (_Float16)0.0f;
}

// ---------------------------------------------------------------- gather sliced layers + LayerNorm -> f16
// one block per (b,n,j); also emits raw layer-24 row (j==0) for the input GEMM
__global__ __launch_bounds__(256) void k_gather_ln(const float* __restrict__ feats,
                                                   const int* __restrict__ idx,
                                                   const float* __restrict__ gamma,
                                                   const float* __restrict__ beta,
                                                   _Float16* __restrict__ XnH,
                                                   _Float16* __restrict__ FinpH) {
  int bid = blockIdx.x;
  int j = bid % JJ;
  int n = (bid / JJ) % HWN;
  int b = bid / (JJ * HWN);
  int r = n / HP, c = n % HP;
  int tok = idx[b] + 1 + r * (HP + 1) + c;
  int layer = 24 - 4 * j;                       // LAYER_IDX[j]
  const float* src = feats + (((long long)b * NTOK + tok) * LLAY + layer) * DDIM;

  float vals[8];
  float lsum = 0.f, lsq = 0.f;
  for (int i = 0; i < 8; i++) {
    float v = src[threadIdx.x + 256 * i];
    vals[i] = v; lsum += v; lsq += v * v;
  }
  for (int o = 16; o > 0; o >>= 1) { lsum += __shfl_xor(lsum, o); lsq += __shfl_xor(lsq, o); }
  __shared__ float ssum[8], ssq[8];
  int wv = threadIdx.x >> 5, ln = threadIdx.x & 31;
  if (ln == 0) { ssum[wv] = lsum; ssq[wv] = lsq; }
  __syncthreads();
  if (wv == 0) {
    float a = (ln < 8) ? ssum[ln] : 0.f;
    float q = (ln < 8) ? ssq[ln] : 0.f;
    for (int o = 4; o > 0; o >>= 1) { a += __shfl_xor(a, o); q += __shfl_xor(q, o); }
    if (ln == 0) { ssum[0] = a; ssq[0] = q; }
  }
  __syncthreads();
  float mu   = ssum[0] * (1.0f / DDIM);
  float var  = ssq[0] * (1.0f / DDIM) - mu * mu;
  float rstd = rsqrtf(var + LN_EPS);

  const float* g  = gamma + (long long)j * DDIM;
  const float* be = beta  + (long long)j * DDIM;
  _Float16* xdst = XnH + (((long long)b * JJ + j) * HWN + n) * DDIM;
  for (int i = 0; i < 8; i++) {
    int d = threadIdx.x + 256 * i;
    xdst[d] = (_Float16)((vals[i] - mu) * rstd * g[d] + be[d]);
  }
  if (j == 0) {  // layer 24 == f[:, -1]
    _Float16* fd = FinpH + ((long long)b * HWN + n) * DDIM;
    for (int i = 0; i < 8; i++) fd[threadIdx.x + 256 * i] = (_Float16)vals[i];
  }
}

// ---------------------------------------------------------------- generic f16 WMMA GEMM
// D[z] = A[z](MxK) * Bt[z%modB](NxK)^T  (+ C[z]) (+ bias)   (f32 out)
// one wave owns a 16(M) x 64(N) strip: 4 accumulators, A-fragment reused 4x.
__global__ __launch_bounds__(256) void k_gemm_f16(const _Float16* __restrict__ A, long long sA,
                                                  const _Float16* __restrict__ Bt, long long sB, int modB,
                                                  const float* __restrict__ C, long long sC,
                                                  const float* __restrict__ bias,
                                                  float* __restrict__ D, long long sD,
                                                  int M, int N, int K) {
  int z = blockIdx.z;
  const _Float16* Ab = A + (long long)z * sA;
  const _Float16* Bb = Bt + (long long)(z % modB) * sB;
  float* Db = D + (long long)z * sD;

  int ngroups = N >> 6;            // 64-wide N strips
  int mtiles  = M >> 4;
  int w = blockIdx.x * (blockDim.x >> 5) + (threadIdx.x >> 5);
  if (w >= mtiles * ngroups) return;
  int mt = w / ngroups, ng = w % ngroups;

  int lane = threadIdx.x & 31;
  int l16 = lane & 15, lhi = lane >> 4;

  // A 16x32 f16 fragment layout: lane<16 -> K 0..15; lane>=16 -> K 16..31
  const _Float16* arow  = Ab + (long long)(mt * 16 + l16) * K + lhi * 16;
  // B 32x16 f16 fragment layout: lanes0-15 -> K 0..15; lanes16-31 -> K 16..31; col = lane&15
  const _Float16* brow0 = Bb + (long long)(ng * 64 + l16) * K + lhi * 16;
  const long long bstep = 16LL * K;

  v8f acc[4];
  if (C) {
    const float* Cb = C + (long long)z * sC;
    for (int t = 0; t < 4; t++)
      for (int r = 0; r < 8; r++) {
        int m = mt * 16 + lhi * 8 + r;
        int n = ng * 64 + t * 16 + l16;
        acc[t][r] = Cb[(long long)m * N + n];
      }
  } else {
    for (int t = 0; t < 4; t++)
      for (int r = 0; r < 8; r++) acc[t][r] = 0.0f;
  }

  for (int k0 = 0; k0 < K; k0 += 32) {
    v8h alo = *(const v8h*)(arow + k0);
    v8h ahi = *(const v8h*)(arow + k0 + 8);
    v16h af;
#pragma unroll
    for (int i = 0; i < 8; i++) { af[i] = alo[i]; af[i + 8] = ahi[i]; }
#pragma unroll
    for (int t = 0; t < 4; t++) {
      v16h bf = *(const v16h*)(brow0 + (long long)t * bstep + k0);
      acc[t] = __builtin_amdgcn_wmma_f32_16x16x32_f16(
          false, af, false, bf, (short)0, acc[t], false, false);
    }
  }

  for (int t = 0; t < 4; t++) {
    int n = ng * 64 + t * 16 + l16;
    float bv = bias ? bias[n] : 0.0f;
    for (int r = 0; r < 8; r++) {
      int m = mt * 16 + lhi * 8 + r;
      Db[(long long)m * N + n] = acc[t][r] + bv;
    }
  }
}

// ---------------------------------------------------------------- 8-head / 6-key attention fuse
// block per (b,n); wave per head; lane owns 4 contiguous dims of 128.
__global__ __launch_bounds__(256) void k_attention(const float* __restrict__ vis,
                                                   _Float16* __restrict__ fuseH) {
  int n = blockIdx.x % HWN;
  int b = blockIdx.x / HWN;
  int h = threadIdx.x >> 5;
  int lane = threadIdx.x & 31;
  const long long js = (long long)HWN * CLIPC;
  const float* base = vis + (((long long)b * JJ) * HWN + n) * CLIPC;
  int d0 = h * 128 + lane * 4;

  float4 q = *(const float4*)(base + d0);
  float4 kv[6];
  float logit[6];
#pragma unroll
  for (int j = 0; j < 6; j++) {
    float4 k = *(const float4*)(base + (long long)(j + 1) * js + d0);
    kv[j] = k;
    float p = q.x * k.x + q.y * k.y + q.z * k.z + q.w * k.w;
    for (int o = 16; o > 0; o >>= 1) p += __shfl_xor(p, o);
    logit[j] = p * SCALE;
  }
  float m = logit[0];
#pragma unroll
  for (int j = 1; j < 6; j++) m = fmaxf(m, logit[j]);
  float e[6], s = 0.f;
#pragma unroll
  for (int j = 0; j < 6; j++) { e[j] = __expf(logit[j] - m); s += e[j]; }
  float inv = 1.0f / s;
  float fx = 0.f, fy = 0.f, fz = 0.f, fw = 0.f;
#pragma unroll
  for (int j = 0; j < 6; j++) {
    float a = e[j] * inv;
    fx += a * kv[j].x; fy += a * kv[j].y; fz += a * kv[j].z; fw += a * kv[j].w;
  }
  _Float16* dst = fuseH + ((long long)b * HWN + n) * CLIPC + d0;
  dst[0] = (_Float16)fx; dst[1] = (_Float16)fy; dst[2] = (_Float16)fz; dst[3] = (_Float16)fw;
}

// ---------------------------------------------------------------- per-token cosine
__global__ __launch_bounds__(256) void k_cosine(const float* __restrict__ out,
                                                const float* __restrict__ cf,
                                                float* __restrict__ cosbuf) {
  long long base = (long long)blockIdx.x * CLIPC;
  float d = 0.f, no = 0.f, nc = 0.f;
  for (int i = threadIdx.x; i < CLIPC; i += 256) {
    float o = out[base + i], c = cf[base + i];
    d += o * c; no += o * o; nc += c * c;
  }
  for (int o = 16; o > 0; o >>= 1) {
    d += __shfl_xor(d, o); no += __shfl_xor(no, o); nc += __shfl_xor(nc, o);
  }
  __shared__ float sd[8], sn[8], sc[8];
  int wv = threadIdx.x >> 5, ln = threadIdx.x & 31;
  if (ln == 0) { sd[wv] = d; sn[wv] = no; sc[wv] = nc; }
  __syncthreads();
  if (threadIdx.x == 0) {
    float td = 0.f, tn = 0.f, tc = 0.f;
    for (int i = 0; i < 8; i++) { td += sd[i]; tn += sn[i]; tc += sc[i]; }
    cosbuf[blockIdx.x] = td / fmaxf(sqrtf(tn) * sqrtf(tc), COS_EPS);
  }
}

// ---------------------------------------------------------------- final loss reduction
__global__ __launch_bounds__(256) void k_reduce_loss(const float* __restrict__ cosbuf,
                                                     float* __restrict__ out) {
  float s = 0.f;
  for (int i = threadIdx.x; i < BQ * HWN; i += 256) s += cosbuf[i];
  for (int o = 16; o > 0; o >>= 1) s += __shfl_xor(s, o);
  __shared__ float sw[8];
  int wv = threadIdx.x >> 5, ln = threadIdx.x & 31;
  if (ln == 0) sw[wv] = s;
  __syncthreads();
  if (threadIdx.x == 0) {
    float t = 0.f;
    for (int i = 0; i < 8; i++) t += sw[i];
    out[0] = 1.0f - t / (float)(BQ * HWN);
  }
}

// ================================================================ host
extern "C" void kernel_launch(void* const* d_in, const int* in_sizes, int n_in,
                              void* d_out, int out_size, void* d_ws, size_t ws_size,
                              hipStream_t stream) {
  (void)in_sizes; (void)n_in; (void)out_size; (void)ws_size;

  const int*   input_ids = (const int*)d_in[0];
  const float* images    = (const float*)d_in[1];
  const float* feats     = (const float*)d_in[2];
  const float* gamma     = (const float*)d_in[4];
  const float* beta      = (const float*)d_in[5];
  const float* norm_w    = (const float*)d_in[6];
  const float* input_w   = (const float*)d_in[7];
  const float* input_b   = (const float*)d_in[8];
  const float* output_w  = (const float*)d_in[9];
  const float* output_b  = (const float*)d_in[10];
  const float* clip_w    = (const float*)d_in[11];
  const float* clip_b    = (const float*)d_in[12];
  float* out = (float*)d_out;

  char* ws = (char*)d_ws;
  size_t off = 0;
  auto alloc = [&](size_t bytes) -> char* {
    char* p = ws + off;
    off = (off + bytes + 255) & ~(size_t)255;
    return p;
  };

  int*      idx       = (int*)alloc(BQ * sizeof(int));
  _Float16* patchesH  = (_Float16*)alloc((size_t)BQ * HWN * KPATP * 2);
  _Float16* clip_wH   = (_Float16*)alloc((size_t)CLIPC * KPATP * 2);
  float*    cf        = (float*)alloc((size_t)BQ * HWN * CLIPC * 4);
  _Float16* XnH       = (_Float16*)alloc((size_t)BQ * JJ * HWN * DDIM * 2);
  _Float16* FinpH     = (_Float16*)alloc((size_t)BQ * HWN * DDIM * 2);
  _Float16* norm_wH   = (_Float16*)alloc((size_t)JJ * CLIPC * DDIM * 2);
  _Float16* input_wH  = (_Float16*)alloc((size_t)CLIPC * DDIM * 2);
  _Float16* output_wH = (_Float16*)alloc((size_t)CLIPC * CLIPC * 2);
  float*    visbuf    = (float*)alloc((size_t)BQ * JJ * HWN * CLIPC * 4);
  float*    inpbuf    = (float*)alloc((size_t)BQ * HWN * CLIPC * 4);
  _Float16* fuseH     = (_Float16*)alloc((size_t)BQ * HWN * CLIPC * 2);
  float*    outbuf    = (float*)alloc((size_t)BQ * HWN * CLIPC * 4);
  float*    cosbuf    = (float*)alloc((size_t)BQ * HWN * 4);

  auto nblk = [](long long total) { return (unsigned)((total + 255) / 256); };

  // stage 0: image-token positions
  k_find_idx<<<1, 256, 0, stream>>>(input_ids, idx);

  // stage 1: f16 staging (patches + weights)
  k_patchify<<<nblk((long long)BQ * HWN * KPATP), 256, 0, stream>>>(images, patchesH);
  k_cvt_pad<<<nblk((long long)CLIPC * KPATP), 256, 0, stream>>>(clip_w, clip_wH, CLIPC, KPAT, KPATP);
  k_cvt_pad<<<nblk((long long)JJ * CLIPC * DDIM), 256, 0, stream>>>(norm_w, norm_wH, (long long)JJ * CLIPC, DDIM, DDIM);
  k_cvt_pad<<<nblk((long long)CLIPC * DDIM), 256, 0, stream>>>(input_w, input_wH, CLIPC, DDIM, DDIM);
  k_cvt_pad<<<nblk((long long)CLIPC * CLIPC), 256, 0, stream>>>(output_w, output_wH, CLIPC, CLIPC, CLIPC);

  // stage 2: gather + LayerNorm
  k_gather_ln<<<BQ * HWN * JJ, 256, 0, stream>>>(feats, idx, gamma, beta, XnH, FinpH);

  // wave-tile grid: (576/16)*(1024/64) = 576 waves -> 72 blocks of 8 waves
  dim3 gblk(72, 1, 1);

  // stage 3: clip_feats = patches @ clip_w.T + clip_b
  gblk.z = BQ;
  k_gemm_f16<<<gblk, 256, 0, stream>>>(patchesH, (long long)HWN * KPATP,
                                       clip_wH, 0, 1,
                                       nullptr, 0, clip_b,
                                       cf, (long long)HWN * CLIPC,
                                       HWN, CLIPC, KPATP);

  // stage 4: vis[b,j] = xn[b,j] @ norm_w[j].T   (z = b*7+j)
  gblk.z = BQ * JJ;
  k_gemm_f16<<<gblk, 256, 0, stream>>>(XnH, (long long)HWN * DDIM,
                                       norm_wH, (long long)CLIPC * DDIM, JJ,
                                       nullptr, 0, nullptr,
                                       visbuf, (long long)HWN * CLIPC,
                                       HWN, CLIPC, DDIM);

  // stage 5: inp = f[:,-1] @ input_w.T + input_b
  gblk.z = BQ;
  k_gemm_f16<<<gblk, 256, 0, stream>>>(FinpH, (long long)HWN * DDIM,
                                       input_wH, 0, 1,
                                       nullptr, 0, input_b,
                                       inpbuf, (long long)HWN * CLIPC,
                                       HWN, CLIPC, DDIM);

  // stage 6: attention fuse
  k_attention<<<BQ * HWN, 256, 0, stream>>>(visbuf, fuseH);

  // stage 7: out = inp + fuse @ output_w.T + output_b
  gblk.z = BQ;
  k_gemm_f16<<<gblk, 256, 0, stream>>>(fuseH, (long long)HWN * CLIPC,
                                       output_wH, 0, 1,
                                       inpbuf, (long long)HWN * CLIPC, output_b,
                                       outbuf, (long long)HWN * CLIPC,
                                       HWN, CLIPC, CLIPC);

  // stage 8: cosine + loss
  k_cosine<<<BQ * HWN, 256, 0, stream>>>(outbuf, cf, cosbuf);
  k_reduce_loss<<<1, 256, 0, stream>>>(cosbuf, out);
}
